// MultiCommodityTransformer_8959301780105
// MI455X (gfx1250) — matrix operand order repaired and checked by reading, not compiled, not run on hardware
//
#include <hip/hip_runtime.h>
#include <hip/hip_bf16.h>

typedef __attribute__((ext_vector_type(16))) _Float16 v16h;
typedef __attribute__((ext_vector_type(8)))  float    v8f;

#define B_  2048
#define C_  64
#define D_  256
#define H_  8
#define HD_ 32
#define F_  1024

// ---------------------------------------------------------------------------
// WMMA fragment loaders (wave32).  ISA 7.12.2 layouts:
//  A (16x32 f16): lane L holds row M=L&15; lanes 0-15 get K={0..7,16..23},
//                 lanes 16-31 get K={8..15,24..31} (2 halves per VGPR).
//  B (32x16 f16): lane L holds col N=L&15; lanes 0-15 K={0..7,16..23}, lanes 16-31
//                 K={8..15,24..31}.
//  C/D (16x16 f32): VGPR r: M = r + 8*(lane>=16), N = lane&15.
// ---------------------------------------------------------------------------
__device__ inline v16h load_a_frag(const _Float16* base, int ld) {
  const int lane = threadIdx.x & 31;
  const int row  = lane & 15;
  const int kb   = (lane >> 4) << 3;          // 0 or 8
  const _Float16* p = base + row * ld + kb;
  v16h a;
#pragma unroll
  for (int e = 0; e < 8; ++e) a[e] = p[e];          // K = kb+0..7
#pragma unroll
  for (int e = 0; e < 8; ++e) a[8 + e] = p[16 + e]; // K = kb+16..23
  return a;
}

// B stored transposed in LDS: baseT[n][k] with leading dim ld (halves).
__device__ inline v16h load_b_frag_t(const _Float16* baseT, int ld) {
  const int lane = threadIdx.x & 31;
  const int col  = lane & 15;
  const int kb   = (lane >> 4) << 3;          // 0 or 8
  const _Float16* p = baseT + col * ld + kb;
  v16h b;
#pragma unroll
  for (int e = 0; e < 8; ++e) b[e] = p[e];
#pragma unroll
  for (int e = 0; e < 8; ++e) b[8 + e] = p[16 + e];
  return b;
}

// ---------------------------------------------------------------------------
// gfx1250 async global->LDS copy (16B per issue), tracked by ASYNCcnt.
// Generic shared pointer low 32 bits == LDS byte offset (flat LDS aperture).
// ---------------------------------------------------------------------------
__device__ inline void async_copy_b128(void* ldsptr, const void* gptr) {
  uint32_t loff = (uint32_t)(uintptr_t)ldsptr;
  uint64_t ga   = (uint64_t)(uintptr_t)gptr;
  asm volatile("global_load_async_to_lds_b128 %0, %1, off"
               :: "v"(loff), "v"(ga) : "memory");
}
__device__ inline void wait_async() {
  asm volatile("s_wait_asynccnt 0x0" ::: "memory");
}

// ---------------------------------------------------------------------------
// Weight conversion: f32 [K,N]  ->  f16 transposed [N,K]  (so GEMM B tiles
// are contiguous 16B-chunk copies for the async-LDS path).
// ---------------------------------------------------------------------------
__global__ void convert_transpose_kernel(const float* __restrict__ in,
                                         _Float16* __restrict__ out,
                                         int K, int N) {
  int i = blockIdx.x * blockDim.x + threadIdx.x;
  if (i < K * N) {
    int k = i / N, n = i % N;
    out[(size_t)n * K + k] = (_Float16)in[i];
  }
}

// ---------------------------------------------------------------------------
// WMMA GEMM:  Y = epilogue(X[MxK] @ W + bias),  W given transposed: WT[N,K] f16.
// 256 threads = 8 waves; block tile 128x128; wave tile 32x64 (8 WMMA/K-step).
// B tile staged via async global->LDS; A tile async when X is f16, VALU
// convert when X is f32.
// ---------------------------------------------------------------------------
template <typename TIN, bool GELU_ACT, bool STORE_HALF>
__global__ __launch_bounds__(256) void wmma_gemm(
    const TIN* __restrict__ X, const _Float16* __restrict__ WT,
    const float* __restrict__ bias, void* __restrict__ Yv,
    int M, int N, int K) {
  __shared__ alignas(16) _Float16 As [128][40];  // [m][k] (+8 pad: 80B rows)
  __shared__ alignas(16) _Float16 BsT[128][40];  // [n][k]

  const int tid  = threadIdx.x;
  const int wave = tid >> 5;
  const int lane = tid & 31;
  const int wrow = wave & 3;        // 4 M-strips of 32 rows
  const int wcol = wave >> 2;       // 2 N-strips of 64 cols
  const int m0 = blockIdx.y * 128;
  const int n0 = blockIdx.x * 128;

  v8f acc[2][4] = {};

  for (int k0 = 0; k0 < K; k0 += 32) {
    // ---- stage A tile 128x32 ----
    if constexpr (sizeof(TIN) == 2) {
      // f16 activation: straight async copies (4 x 16B chunks per row)
#pragma unroll
      for (int i = 0; i < 2; ++i) {
        int id = i * 256 + tid;
        int r = id >> 2, c = (id & 3) * 8;
        async_copy_b128(&As[r][c], X + (size_t)(m0 + r) * K + k0 + c);
      }
    } else {
#pragma unroll
      for (int i = 0; i < 16; ++i) {
        int idx = i * 256 + tid;
        int r = idx >> 5, c = idx & 31;
        As[r][c] = (_Float16)(float)X[(size_t)(m0 + r) * K + k0 + c];
      }
    }
    // ---- stage B tile 128(n) x 32(k), async from transposed weights ----
#pragma unroll
    for (int i = 0; i < 2; ++i) {
      int id = i * 256 + tid;
      int n = id >> 2, c = (id & 3) * 8;
      async_copy_b128(&BsT[n][c], WT + (size_t)(n0 + n) * K + k0 + c);
    }
    wait_async();
    __syncthreads();

    v16h a0 = load_a_frag(&As[wrow * 32][0], 40);
    v16h a1 = load_a_frag(&As[wrow * 32 + 16][0], 40);
#pragma unroll
    for (int nt = 0; nt < 4; ++nt) {
      v16h b = load_b_frag_t(&BsT[wcol * 64 + nt * 16][0], 40);
      acc[0][nt] = __builtin_amdgcn_wmma_f32_16x16x32_f16(
          false, a0, false, b, (short)0, acc[0][nt], false, false);
      acc[1][nt] = __builtin_amdgcn_wmma_f32_16x16x32_f16(
          false, a1, false, b, (short)0, acc[1][nt], false, false);
    }
    __syncthreads();
  }

  const int nlane = lane & 15;
  const int rbase = (lane >> 4) << 3;
#pragma unroll
  for (int half = 0; half < 2; ++half) {
#pragma unroll
    for (int nt = 0; nt < 4; ++nt) {
      int col = n0 + wcol * 64 + nt * 16 + nlane;
      float bcol = bias[col];
#pragma unroll
      for (int r = 0; r < 8; ++r) {
        size_t row = (size_t)(m0 + wrow * 32 + half * 16 + rbase + r);
        float v = acc[half][nt][r] + bcol;
        if constexpr (GELU_ACT)
          v = 0.5f * v * (1.0f + erff(v * 0.70710678118654752f)); // exact GELU
        if constexpr (STORE_HALF)
          ((_Float16*)Yv)[row * N + col] = (_Float16)v;
        else
          ((float*)Yv)[row * N + col] = v;
      }
    }
  }
}

// ---------------------------------------------------------------------------
// Attention: one block per (batch, head).  scores -> masked softmax + gate ->
// attn tensor output -> ctx = attn @ v.  q/k staged via async-LDS.
// ---------------------------------------------------------------------------
__global__ __launch_bounds__(128) void attn_kernel(
    const _Float16* __restrict__ qh, const _Float16* __restrict__ kh,
    const _Float16* __restrict__ vh, const unsigned char* __restrict__ adj,
    const float* __restrict__ head_gate, _Float16* __restrict__ ctx,
    float* __restrict__ attn_out) {
  __shared__ alignas(16) _Float16 qs[64][40];   // [j][d]
  __shared__ alignas(16) _Float16 ks[64][40];   // [j][d]  (B^T for scores)
  __shared__ alignas(16) _Float16 vsT[32][72];  // [d][j]  (B^T for ctx)
  __shared__ float    sc[64][65];               // scores
  __shared__ alignas(16) _Float16 at[64][72];   // attn (A operand for ctx)

  const int tid  = threadIdx.x;
  const int wave = tid >> 5;
  const int lane = tid & 31;
  const int bh = blockIdx.x;
  const int b  = bh >> 3;
  const int h  = bh & 7;
  const size_t base = ((size_t)b * C_) * D_ + (size_t)h * HD_;

  // async-stage q,k rows (64 rows x 4 chunks of 16B)
#pragma unroll
  for (int i = 0; i < 2; ++i) {
    int id = i * 128 + tid;
    int j = id >> 2, c = (id & 3) * 8;
    async_copy_b128(&qs[j][c], qh + base + (size_t)j * D_ + c);
    async_copy_b128(&ks[j][c], kh + base + (size_t)j * D_ + c);
  }
  // v transposed via VALU (transpose not expressible as straight copy)
#pragma unroll
  for (int i = 0; i < 16; ++i) {
    int idx = i * 128 + tid;
    int j = idx >> 5, d = idx & 31;
    vsT[d][j] = vh[base + (size_t)j * D_ + d];
  }
  wait_async();
  __syncthreads();

  { // scores = q @ k^T * 1/sqrt(HD); each wave: rows wave*16..+15, 64 cols
    v16h a = load_a_frag(&qs[wave * 16][0], 40);
    const float scale = 0.17677669529663687f; // 1/sqrt(32)
    const int mloc  = wave * 16 + ((lane >> 4) << 3);
    const int nlane = lane & 15;
#pragma unroll
    for (int nt = 0; nt < 4; ++nt) {
      v16h bfr = load_b_frag_t(&ks[nt * 16][0], 40);
      v8f acc = {};
      acc = __builtin_amdgcn_wmma_f32_16x16x32_f16(
          false, a, false, bfr, (short)0, acc, false, false);
#pragma unroll
      for (int r = 0; r < 8; ++r) sc[mloc + r][nt * 16 + nlane] = acc[r] * scale;
    }
  }
  __syncthreads();

  if (tid < 64) {                   // softmax row i = tid
    const int i = tid;
    float hg[H_], gm = -1e30f;
#pragma unroll
    for (int hh = 0; hh < H_; ++hh) { hg[hh] = head_gate[i * H_ + hh]; gm = fmaxf(gm, hg[hh]); }
    float gs = 0.f;
#pragma unroll
    for (int hh = 0; hh < H_; ++hh) gs += __expf(hg[hh] - gm);
    const float gate = __expf(hg[h] - gm) / gs;

    float mx = -1e30f;
    for (int j = 0; j < 64; ++j) {
      float s = adj[i * C_ + j] ? sc[i][j] : -1e30f;
      sc[i][j] = s;
      mx = fmaxf(mx, s);
    }
    float sum = 0.f;
    for (int j = 0; j < 64; ++j) {
      float p = __expf(sc[i][j] - mx);
      sc[i][j] = p;
      sum += p;
    }
    // attn = softmax * gate, renormalized by (sum_j attn + 1e-8)
    const float outscale = gate / (sum * (gate + 1e-8f));
    const size_t obase = (((size_t)b * C_ + i) * H_ + h) * C_;
    for (int j = 0; j < 64; ++j) {
      float o = sc[i][j] * outscale;
      at[i][j] = (_Float16)o;
      attn_out[obase + j] = o;   // [B, C, H, 1, C]
    }
  }
  __syncthreads();

  { // ctx = attn[64x64] @ v[64x32]
    v8f acc[2] = {};
#pragma unroll
    for (int k0 = 0; k0 < 64; k0 += 32) {
      v16h a = load_a_frag(&at[wave * 16][k0], 72);
#pragma unroll
      for (int nt = 0; nt < 2; ++nt) {
        v16h bfr = load_b_frag_t(&vsT[nt * 16][k0], 72);
        acc[nt] = __builtin_amdgcn_wmma_f32_16x16x32_f16(
            false, a, false, bfr, (short)0, acc[nt], false, false);
      }
    }
    const int mloc = wave * 16 + ((lane >> 4) << 3);
#pragma unroll
    for (int nt = 0; nt < 2; ++nt) {
      int d = nt * 16 + (lane & 15);
#pragma unroll
      for (int r = 0; r < 8; ++r)
        ctx[base + (size_t)(mloc + r) * D_ + d] = (_Float16)acc[nt][r];
    }
  }
}

// ---------------------------------------------------------------------------
// out = LayerNorm(x + y) * w + b ; one 256-thread block per row (D=256)
// ---------------------------------------------------------------------------
__global__ __launch_bounds__(256) void add_ln_kernel(
    const float* __restrict__ x, const float* __restrict__ y,
    const float* __restrict__ w, const float* __restrict__ bsc,
    float* __restrict__ out) {
  __shared__ float red[256];
  const int t = threadIdx.x;
  const size_t row = blockIdx.x;
  float v = x[row * D_ + t] + y[row * D_ + t];

  red[t] = v; __syncthreads();
  for (int s = 128; s > 0; s >>= 1) { if (t < s) red[t] += red[t + s]; __syncthreads(); }
  const float mean = red[0] * (1.0f / D_);
  __syncthreads();

  const float d = v - mean;
  red[t] = d * d; __syncthreads();
  for (int s = 128; s > 0; s >>= 1) { if (t < s) red[t] += red[t + s]; __syncthreads(); }
  const float var = red[0] * (1.0f / D_);

  out[row * D_ + t] = d * rsqrtf(var + 1e-5f) * w[t] + bsc[t];
}

// ---------------------------------------------------------------------------
extern "C" void kernel_launch(void* const* d_in, const int* in_sizes, int n_in,
                              void* d_out, int out_size, void* d_ws, size_t ws_size,
                              hipStream_t stream) {
  const size_t M = (size_t)B_ * C_;  // 131072 rows

  const float*         states = (const float*)d_in[0];
  const unsigned char* adj    = (const unsigned char*)d_in[1];  // jax bool = 1 byte
  const float* Wq = (const float*)d_in[2];
  const float* bq = (const float*)d_in[3];
  const float* Wk = (const float*)d_in[4];
  const float* bk = (const float*)d_in[5];
  const float* Wv = (const float*)d_in[6];
  const float* bv = (const float*)d_in[7];
  const float* Wo = (const float*)d_in[8];
  const float* bo = (const float*)d_in[9];
  const float* head_gate = (const float*)d_in[10];
  const float* ln1_w = (const float*)d_in[11];
  const float* ln1_b = (const float*)d_in[12];
  const float* ln2_w = (const float*)d_in[13];
  const float* ln2_b = (const float*)d_in[14];
  const float* W1 = (const float*)d_in[15];
  const float* b1 = (const float*)d_in[16];
  const float* W2 = (const float*)d_in[17];
  const float* b2 = (const float*)d_in[18];

  float* updated  = (float*)d_out;               // [B,C,D]
  float* attn_out = (float*)d_out + M * D_;      // [B,C,H,1,C]

  // ---- workspace carve-up ----
  char* ws = (char*)d_ws;
  size_t off = 0;
  auto carve = [&](size_t bytes) -> char* {
    char* p = ws + off;
    off += (bytes + 255) & ~(size_t)255;
    return p;
  };
  _Float16* qh   = (_Float16*)carve(M * D_ * sizeof(_Float16));
  _Float16* khb  = (_Float16*)carve(M * D_ * sizeof(_Float16));
  _Float16* vhb  = (_Float16*)carve(M * D_ * sizeof(_Float16));
  _Float16* ctxh = (_Float16*)carve(M * D_ * sizeof(_Float16));
  _Float16* WqT  = (_Float16*)carve((size_t)D_ * D_ * 2);   // all weights stored [N,K] f16
  _Float16* WkT  = (_Float16*)carve((size_t)D_ * D_ * 2);
  _Float16* WvT  = (_Float16*)carve((size_t)D_ * D_ * 2);
  _Float16* WoT  = (_Float16*)carve((size_t)D_ * D_ * 2);
  _Float16* W1T  = (_Float16*)carve((size_t)D_ * F_ * 2);
  _Float16* W2T  = (_Float16*)carve((size_t)F_ * D_ * 2);
  float*    tmp   = (float*)carve(M * D_ * sizeof(float));  // O-proj / FFN2 out
  float*    resid = (float*)carve(M * D_ * sizeof(float));
  _Float16* h1    = (_Float16*)carve(M * F_ * sizeof(_Float16));

  // ---- weight conversion + transpose (cheap, per-call for determinism) ----
  convert_transpose_kernel<<<(D_ * D_ + 255) / 256, 256, 0, stream>>>(Wq, WqT, D_, D_);
  convert_transpose_kernel<<<(D_ * D_ + 255) / 256, 256, 0, stream>>>(Wk, WkT, D_, D_);
  convert_transpose_kernel<<<(D_ * D_ + 255) / 256, 256, 0, stream>>>(Wv, WvT, D_, D_);
  convert_transpose_kernel<<<(D_ * D_ + 255) / 256, 256, 0, stream>>>(Wo, WoT, D_, D_);
  convert_transpose_kernel<<<(D_ * F_ + 255) / 256, 256, 0, stream>>>(W1, W1T, D_, F_);
  convert_transpose_kernel<<<(F_ * D_ + 255) / 256, 256, 0, stream>>>(W2, W2T, F_, D_);

  const dim3 blk(256);
  const dim3 gD(D_ / 128, M / 128);   // (2, 1024)
  const dim3 gF(F_ / 128, M / 128);   // (8, 1024)

  // ---- Q, K, V projections (store f16) ----
  wmma_gemm<float, false, true><<<gD, blk, 0, stream>>>(states, WqT, bq, qh,  (int)M, D_, D_);
  wmma_gemm<float, false, true><<<gD, blk, 0, stream>>>(states, WkT, bk, khb, (int)M, D_, D_);
  wmma_gemm<float, false, true><<<gD, blk, 0, stream>>>(states, WvT, bv, vhb, (int)M, D_, D_);

  // ---- attention (scores + softmax/gate + ctx), writes attn tensor output ----
  attn_kernel<<<B_ * H_, 128, 0, stream>>>(qh, khb, vhb, adj, head_gate, ctxh, attn_out);

  // ---- output projection + LN1 ----
  wmma_gemm<_Float16, false, false><<<gD, blk, 0, stream>>>(ctxh, WoT, bo, tmp, (int)M, D_, D_);
  add_ln_kernel<<<M, 256, 0, stream>>>(states, tmp, ln1_w, ln1_b, resid);

  // ---- FFN: GELU(resid @ W1 + b1) @ W2 + b2, then LN2 -> updated ----
  wmma_gemm<float, true, true><<<gF, blk, 0, stream>>>(resid, W1T, b1, h1, (int)M, F_, D_);
  wmma_gemm<_Float16, false, false><<<gD, blk, 0, stream>>>(h1, W2T, b2, tmp, (int)M, D_, F_);
  add_ln_kernel<<<M, 256, 0, stream>>>(resid, tmp, ln2_w, ln2_b, updated);
}
